// CharRNN_40312563040738
// MI455X (gfx1250) — matrix operand order repaired and checked by reading, not compiled, not run on hardware
//
#include <hip/hip_runtime.h>
#include <hip/hip_bf16.h>

typedef __attribute__((ext_vector_type(16))) _Float16 v16h;
typedef __attribute__((ext_vector_type(8)))  _Float16 v8h;
typedef __attribute__((ext_vector_type(8)))  float    v8f;

#define B_  64
#define S_  256
#define T_  16
#define V_  262
#define E_  64
#define H_  32
#define NTILES 18   // 12 Wih B-tiles (6 gate tiles x 2 K-chunks) + 6 Whh B-tiles

// ---------- helpers ----------

__device__ __forceinline__ v8f splat8(float s) {
    v8f r;
#pragma unroll
    for (int i = 0; i < 8; ++i) r[i] = s;
    return r;
}

// Load 16 halves in WMMA A-layout order for this lane from a row-major row:
// lanes 0-15 want K = {base..base+7, base+16..base+23}; caller passes row+lh*8.
__device__ __forceinline__ v16h ldA16(const _Float16* row) {
    v8h lo = *(const v8h*)(row);
    v8h hi = *(const v8h*)(row + 16);
    return __builtin_shufflevector(lo, hi, 0,1,2,3,4,5,6,7,8,9,10,11,12,13,14,15);
}

// B-layout tile: pre-permuted in LDS as [tile][lane][8 x packed-f16-pair]
__device__ __forceinline__ v16h ldB16(const uint32_t* s_wB, int tile, int lane) {
    return *(const v16h*)(s_wB + (tile * 32 + lane) * 8);
}

__device__ __forceinline__ v8f wmma_f16(v16h a, v16h b, v8f c) {
    return __builtin_amdgcn_wmma_f32_16x16x32_f16(
        /*neg_a=*/false, a, /*neg_b=*/false, b,
        /*c_mod=*/(short)0, c, /*reuse_a=*/false, /*reuse_b=*/false);
}

// Fast activations: hardware-rate TRANS ops (v_exp_f32 / v_rcp_f32 / v_tanh_f32),
// avoiding the precise IEEE divide expansion and ocml tanh.
__device__ __forceinline__ float fast_sigmoid(float x) {
    return __builtin_amdgcn_rcpf(1.f + __expf(-x));
}

__device__ __forceinline__ float fast_tanh(float x) {
#if __has_builtin(__builtin_amdgcn_tanhf)
    return __builtin_amdgcn_tanhf(x);          // v_tanh_f32 (gfx1250 TRANS op)
#else
    // tanh(x) = 2*sigmoid(2x) - 1, via v_exp + v_rcp
    float s = __builtin_amdgcn_rcpf(1.f + __expf(-2.f * x));
    return __builtin_fmaf(2.f, s, -1.f);
#endif
}

// Permute fp32 weights into packed-f16 B-layout tiles in LDS.
// Tile t<12: Wih, gate-tile g=t/2 (channels g*16..), K-chunk kc=t&1 (E 32*kc..).
// Tile t>=12: Whh, gate-tile g=t-12, K=hidden (32).
// B layout (K=32, N=16): lanes 0-15 hold K 0..7,16..23 (pairs per slot), lanes 16-31 K 8..15,24..31.
__device__ __forceinline__ void convert_weights(uint32_t* s_wB,
                                                const float* __restrict__ Wih,
                                                const float* __restrict__ Whh,
                                                int tid) {
    for (int w = tid; w < NTILES * 256; w += 256) {
        int tile = w >> 8;
        int rem  = w & 255;
        int ln   = rem >> 3;      // lane 0..31
        int v    = rem & 7;       // slot 0..7
        int nc    = ln & 15;                      // N within tile
        int khalf = ((ln >> 4) << 3) + 2 * v + ((v >> 2) << 3);     // K within 32-chunk
        float a, b;
        if (tile < 12) {
            int g = tile >> 1, kc = tile & 1;
            int N = g * 16 + nc;
            int K = kc * 32 + khalf;
            a = Wih[N * E_ + K];
            b = Wih[N * E_ + K + 1];
        } else {
            int g = tile - 12;
            int N = g * 16 + nc;
            a = Whh[N * H_ + khalf];
            b = Whh[N * H_ + khalf + 1];
        }
        union { _Float16 h[2]; uint32_t u; } p;
        p.h[0] = (_Float16)a;
        p.h[1] = (_Float16)b;
        s_wB[w] = p.u;
    }
}

// One GRU direction for this wave's 16 words; writes temporal max to out channels [cbase, cbase+32).
__device__ __forceinline__ void run_dir(const int* __restrict__ x, int n0, int lane,
                                        const _Float16* s_emb, const uint32_t* s_wB,
                                        _Float16* s_hw,
                                        const float* __restrict__ bih,
                                        const float* __restrict__ bhh,
                                        float* __restrict__ out, int cbase, bool reverse) {
    const int lc = lane & 15;   // N-column (channel) / word for token fetch
    const int lh = lane >> 4;   // lane-half

    // zero h scratch (wave-local, DScnt keeps order)
    for (int i = lane; i < 16 * H_; i += 32) s_hw[i] = (_Float16)0.f;

    // per-lane bias values (channel = lc within each 16-wide gate tile)
    const float bs_r0 = bih[lc]        + bhh[lc];
    const float bs_r1 = bih[16 + lc]   + bhh[16 + lc];
    const float bs_z0 = bih[32 + lc]   + bhh[32 + lc];
    const float bs_z1 = bih[48 + lc]   + bhh[48 + lc];
    const float bi_n0 = bih[64 + lc],  bi_n1 = bih[80 + lc];
    const float bh_n0 = bhh[64 + lc],  bh_n1 = bhh[80 + lc];

    // Weight tiles: load once, live in VGPRs across the whole time loop.
    v16h wB[NTILES];
#pragma unroll
    for (int i = 0; i < NTILES; ++i) wB[i] = ldB16(s_wB, i, lane);

    v8f h0 = splat8(0.f), h1 = splat8(0.f);         // C-layout h (ch 0..15 / 16..31)
    v8f hx0 = splat8(-3.402823e38f), hx1 = splat8(-3.402823e38f);

    const int word = n0 + lc;

    for (int t = 0; t < T_; ++t) {
        const int tt = reverse ? (T_ - 1 - t) : t;
        const int token = x[word * T_ + tt];

        // A tiles: embedded chars (K=E split in two 32-chunks) and hidden state
        const _Float16* er = s_emb + token * E_ + lh * 8;
        v16h a0 = ldA16(er);
        v16h a1 = ldA16(er + 32);
        v16h ah = ldA16(s_hw + lc * H_ + lh * 8);

        v8f acc_r0 = splat8(bs_r0), acc_r1 = splat8(bs_r1);
        v8f acc_z0 = splat8(bs_z0), acc_z1 = splat8(bs_z1);
        v8f accI0  = splat8(bi_n0), accI1  = splat8(bi_n1);
        v8f accH0  = splat8(bh_n0), accH1  = splat8(bh_n1);

        // gi = xe @ Wih^T  (r/z fused with gh accumulation; n-gate input kept separate)
        acc_r0 = wmma_f16(a0, wB[0],  acc_r0);
        acc_r0 = wmma_f16(a1, wB[1],  acc_r0);
        acc_r1 = wmma_f16(a0, wB[2],  acc_r1);
        acc_r1 = wmma_f16(a1, wB[3],  acc_r1);
        acc_z0 = wmma_f16(a0, wB[4],  acc_z0);
        acc_z0 = wmma_f16(a1, wB[5],  acc_z0);
        acc_z1 = wmma_f16(a0, wB[6],  acc_z1);
        acc_z1 = wmma_f16(a1, wB[7],  acc_z1);
        accI0  = wmma_f16(a0, wB[8],  accI0);
        accI0  = wmma_f16(a1, wB[9],  accI0);
        accI1  = wmma_f16(a0, wB[10], accI1);
        accI1  = wmma_f16(a1, wB[11], accI1);

        // gh = h @ Whh^T  (K = 32 = H, single WMMA per tile)
        acc_r0 = wmma_f16(ah, wB[12], acc_r0);
        acc_r1 = wmma_f16(ah, wB[13], acc_r1);
        acc_z0 = wmma_f16(ah, wB[14], acc_z0);
        acc_z1 = wmma_f16(ah, wB[15], acc_z1);
        accH0  = wmma_f16(ah, wB[16], accH0);
        accH1  = wmma_f16(ah, wB[17], accH1);

        // GRU gates + state update + running temporal max (elementwise in C layout)
#pragma unroll
        for (int e = 0; e < 8; ++e) {
            float r0 = fast_sigmoid(acc_r0[e]);
            float r1 = fast_sigmoid(acc_r1[e]);
            float z0 = fast_sigmoid(acc_z0[e]);
            float z1 = fast_sigmoid(acc_z1[e]);
            float n0g = fast_tanh(__builtin_fmaf(r0, accH0[e], accI0[e]));
            float n1g = fast_tanh(__builtin_fmaf(r1, accH1[e], accI1[e]));
            float hn0 = __builtin_fmaf(z0, h0[e] - n0g, n0g);   // (1-z)*n + z*h
            float hn1 = __builtin_fmaf(z1, h1[e] - n1g, n1g);
            h0[e] = hn0; h1[e] = hn1;
            hx0[e] = fmaxf(hx0[e], hn0);
            hx1[e] = fmaxf(hx1[e], hn1);
        }

        // C-layout -> row-major f16 scratch for next step's A-layout load
#pragma unroll
        for (int p = 0; p < 8; ++p) {
            int w = p + lh * 8;
            s_hw[w * H_ + lc]      = (_Float16)h0[p];
            s_hw[w * H_ + 16 + lc] = (_Float16)h1[p];
        }
    }

    // store temporal max (C layout: VGPR p -> word p (+8 for upper lane-half), channel = lc)
#pragma unroll
    for (int p = 0; p < 8; ++p) {
        int w = n0 + p + lh * 8;
        out[w * (2 * H_) + cbase + lc]      = hx0[p];
        out[w * (2 * H_) + cbase + 16 + lc] = hx1[p];
    }
}

// ---------- kernel ----------

__global__ void __launch_bounds__(256)
charrnn_bigru_wmma_kernel(const int* __restrict__ x,
                          const float* __restrict__ emb,
                          const float* __restrict__ Wih_f, const float* __restrict__ Whh_f,
                          const float* __restrict__ bih_f, const float* __restrict__ bhh_f,
                          const float* __restrict__ Wih_b, const float* __restrict__ Whh_b,
                          const float* __restrict__ bih_b, const float* __restrict__ bhh_b,
                          float* __restrict__ out) {
    __shared__ _Float16 s_emb[V_ * E_];            // 33.5 KB: full embedding table, f16
    __shared__ uint32_t s_wB[NTILES * 32 * 8];     // 18 KB: B-layout weight tiles (one dir)
    __shared__ _Float16 s_h[8][16 * H_];           // 8 KB: per-wave h transpose scratch

    const int tid  = threadIdx.x;
    const int lane = tid & 31;
    const int wv   = tid >> 5;

    // embedding -> LDS as f16 (row 0 is already the zero padding row)
    for (int i = tid; i < V_ * E_; i += 256) s_emb[i] = (_Float16)emb[i];

    convert_weights(s_wB, Wih_f, Whh_f, tid);
    __syncthreads();

    const int n0 = (blockIdx.x * 8 + wv) * 16;   // this wave's first word

    run_dir(x, n0, lane, s_emb, s_wB, s_h[wv], bih_f, bhh_f, out, 0,  false);
    __syncthreads();                              // all waves done with fwd weights
    convert_weights(s_wB, Wih_b, Whh_b, tid);
    __syncthreads();
    run_dir(x, n0, lane, s_emb, s_wB, s_h[wv], bih_b, bhh_b, out, H_, true);
}

// ---------- launch ----------

extern "C" void kernel_launch(void* const* d_in, const int* in_sizes, int n_in,
                              void* d_out, int out_size, void* d_ws, size_t ws_size,
                              hipStream_t stream) {
    (void)in_sizes; (void)n_in; (void)d_ws; (void)ws_size; (void)out_size;
    const int*   x     = (const int*)d_in[0];
    const float* emb   = (const float*)d_in[1];
    const float* Wih_f = (const float*)d_in[2];
    const float* Whh_f = (const float*)d_in[3];
    const float* bih_f = (const float*)d_in[4];
    const float* bhh_f = (const float*)d_in[5];
    const float* Wih_b = (const float*)d_in[6];
    const float* Whh_b = (const float*)d_in[7];
    const float* bih_b = (const float*)d_in[8];
    const float* bhh_b = (const float*)d_in[9];
    float* out = (float*)d_out;

    // 16384 words / (8 waves * 16 words) = 128 blocks of 256 threads (8 wave32s)
    dim3 grid(128), block(256);
    hipLaunchKernelGGL(charrnn_bigru_wmma_kernel, grid, block, 0, stream,
                       x, emb, Wih_f, Whh_f, bih_f, bhh_f,
                       Wih_b, Whh_b, bih_b, bhh_b, out);
}
